// SpatioTemporalCrossAttention_45938970198441
// MI455X (gfx1250) — matrix operand order repaired and checked by reading, not compiled, not run on hardware
//
#include <hip/hip_runtime.h>
#include <hip/hip_bf16.h>
#include <math.h>

// SpatioTemporalCrossAttention — exact (collapsed) semantics for MI455X.
// Shapes: B=256 F=16 FO=32 J=22 H=8 IN=256 OUT=512 HID=64 HS=32.
// The reference einsums are degenerate (see analysis): the whole network reduces
// to 4 skinny GEMMs (WMMA f16->f32), streaming reductions, a 22-wide masked
// softmax diagonal, and a broadcast epilogue. Memory-bound: ~740MB @ 23.3TB/s.

#define B_   256
#define F_   16
#define FO_  32
#define J_   22
#define H_   8
#define IN_  256
#define OUT_ 512
#define HS_  32
#define NEG_ (-1.0e9f)

typedef __attribute__((ext_vector_type(16))) _Float16 v16h;
typedef __attribute__((ext_vector_type(8)))  float    v8f;

// ---------------------------------------------------------------------------
// WMMA GEMM: C(MxN) = cvt_f16(A(MxK f32)) @ WT(NxK f16, pre-transposed)
//            + biasScale * bias[N].   K fixed at 256, fully unrolled -> 8 WMMAs.
// Block = 256 threads = 8 waves arranged WM x WN, each wave owns a 16x16 tile.
// Fragment addressing follows CDNA5 ISA 16-bit layouts (wave32):
//   A 16x32 : lanes 0-15 row M=lane, halves e<8 -> K=e, e>=8 -> K=16+e-8;
//             lanes 16-31 same rows, K offset +8.
//   B 32x16 : lane holds col N=lane&15, same K order as A (lane>=16 -> K+8).
//   C 16x16 : VGPR r -> row r + 8*(lane>=16), col lane&15.
// ---------------------------------------------------------------------------
template<int WM, int WN>
__global__ __launch_bounds__(256)
void gemm16_wmma(const float* __restrict__ A, const _Float16* __restrict__ WT,
                 const float* __restrict__ bias, float biasScale,
                 float* __restrict__ C, int N) {
  constexpr int K = 256;
  const int wave = threadIdx.x >> 5;
  const int lane = threadIdx.x & 31;
  const int wm = wave / WN, wn = wave % WN;
  const int tileM = (blockIdx.x * WM + wm) << 4;
  const int tileN = (blockIdx.y * WN + wn) << 4;
  const int halfSel = lane >> 4;                 // 0: lanes 0-15, 1: lanes 16-31
  const int rowA = tileM + (lane & 15);
  const int colB = tileN + (lane & 15);
  const float*    aBase = A  + (size_t)rowA * K + halfSel * 8;
  const _Float16* bBase = WT + (size_t)colB * K + halfSel * 8;

  v8f acc = {};
#pragma unroll
  for (int k0 = 0; k0 < K; k0 += 32) {
    const float* ap = aBase + k0;
    if (k0 + 32 < K) __builtin_prefetch(ap + 32, 0, 0);   // global_prefetch_b8
    v16h a, b;
#pragma unroll
    for (int e = 0; e < 8; ++e) {
      a[e]     = (_Float16)ap[e];        // K = k0 + 8*halfSel + e
      a[8 + e] = (_Float16)ap[16 + e];   // K = k0 + 8*halfSel + 16 + e
    }
    const _Float16* bp = bBase + k0;
#pragma unroll
    for (int e = 0; e < 8; ++e) { b[e] = bp[e]; b[8 + e] = bp[16 + e]; }
    acc = __builtin_amdgcn_wmma_f32_16x16x32_f16(
        /*neg_a=*/false, a, /*neg_b=*/false, b,
        /*c_mod=*/(short)0, acc, /*reuse_a=*/false, /*reuse_b=*/false);
  }
  const float bb = biasScale * bias[colB];
#pragma unroll
  for (int r = 0; r < 8; ++r) {
    const int row = tileM + r + (halfSel << 3);
    C[(size_t)row * N + colB] = acc[r] + bb;
  }
}

// ---------------------------------------------------------------------------
// Weight prep: fold channel-sums into reduced q/k weights; transpose the
// needed halves of Wv into f16 NxK layout; build bias vectors.
// ---------------------------------------------------------------------------
__global__ __launch_bounds__(256)
void prep_weights(const float* __restrict__ Wq, const float* __restrict__ bq,
                  const float* __restrict__ Wk, const float* __restrict__ bk,
                  const float* __restrict__ Wv, const float* __restrict__ bv,
                  _Float16* __restrict__ WqT, float* __restrict__ biasq,
                  _Float16* __restrict__ WkT, float* __restrict__ biask,
                  _Float16* __restrict__ WvsT, float* __restrict__ bias_vs,
                  _Float16* __restrict__ WvtT, float* __restrict__ bias_vt) {
  const int tid = blockIdx.x * blockDim.x + threadIdx.x;
  const int stride = gridDim.x * blockDim.x;
  // Reduced q/k weights: WqT[n][k] = sum_{c<32} Wq[k][n*64+c]   (n<8, else 0)
  for (int i = tid; i < 16 * 256; i += stride) {
    const int n = i >> 8, k = i & 255;
    float sq = 0.f, sk = 0.f;
    if (n < 8) {
      const float* pq = Wq + (size_t)k * OUT_ + n * 64;
      const float* pk = Wk + (size_t)k * OUT_ + n * 64;
#pragma unroll
      for (int c = 0; c < 32; ++c) { sq += pq[c]; sk += pk[c]; }
    }
    WqT[i] = (_Float16)sq;
    WkT[i] = (_Float16)sk;
  }
  // Wv transposed halves: row r = h*32+c maps to channel h*64+c (s) / +32 (t)
  for (int i = tid; i < 256 * 256; i += stride) {
    const int r = i >> 8, k = i & 255;
    const int h = r >> 5, c = r & 31;
    WvsT[i] = (_Float16)Wv[(size_t)k * OUT_ + h * 64 + c];
    WvtT[i] = (_Float16)Wv[(size_t)k * OUT_ + h * 64 + 32 + c];
  }
  for (int i = tid; i < 16; i += stride) {
    float s1 = 0.f, s2 = 0.f;
    if (i < 8)
      for (int c = 0; c < 32; ++c) { s1 += bq[i * 64 + c]; s2 += bk[i * 64 + c]; }
    biasq[i] = s1; biask[i] = s2;
  }
  for (int i = tid; i < 256; i += stride) {
    const int h = i >> 5, c = i & 31;
    bias_vs[i] = bv[h * 64 + c];
    bias_vt[i] = bv[h * 64 + 32 + c];
  }
}

// vJsum[b*F+f][k] = sum_{u<J} v[b,f,u,k]
__global__ __launch_bounds__(256)
void reduce_v_J(const float* __restrict__ v, float* __restrict__ vJsum) {
  const int idx = blockIdx.x * blockDim.x + threadIdx.x;
  if (idx >= B_ * F_ * IN_) return;
  const int k = idx & 255;
  const int bf = idx >> 8;
  const float* p = v + ((size_t)bf * J_) * IN_ + k;
  float s = 0.f;
#pragma unroll
  for (int u = 0; u < J_; ++u) s += p[u * IN_];
  vJsum[idx] = s;
}

// vFsum[b*J+s][k] = sum_{f<F} v[b,f,s,k]
__global__ __launch_bounds__(256)
void reduce_v_F(const float* __restrict__ v, float* __restrict__ vFsum) {
  const int idx = blockIdx.x * blockDim.x + threadIdx.x;
  if (idx >= B_ * J_ * IN_) return;
  const int k = idx & 255;
  const int bs = idx >> 8;
  const int s = bs % J_, b = bs / J_;
  const float* p = v + (((size_t)b * F_) * J_ + s) * IN_ + k;
  float s0 = 0.f;
#pragma unroll
  for (int f = 0; f < F_; ++f) s0 += p[(size_t)f * J_ * IN_];
  vFsum[idx] = s0;
}

// ku[((b*H+h)*FO+t)*J+u] = sum_F conv_s_w[t,F]*ks2d[((b*F+F)*J+u)*16+h] + 32*conv_s_b[t]
__global__ __launch_bounds__(256)
void explode_ku(const float* __restrict__ ks2d, const float* __restrict__ csw,
                const float* __restrict__ csb, float* __restrict__ ku) {
  const int idx = blockIdx.x * blockDim.x + threadIdx.x;
  if (idx >= B_ * H_ * FO_ * J_) return;
  const int u = idx % J_;
  const int t = (idx / J_) % FO_;
  const int h = (idx / (J_ * FO_)) % H_;
  const int b = idx / (J_ * FO_ * H_);
  float s = 0.f;
#pragma unroll
  for (int f = 0; f < F_; ++f)
    s += csw[t * F_ + f] * ks2d[(((size_t)b * F_ + f) * J_ + u) * 16 + h];
  ku[idx] = s + 32.0f * csb[t];
}

// vssum[((b*H+h)*FO+t)*HS+c] = sum_F conv_s_w[t,F]*vpJ[(b*F+F)*256 + h*32+c] + J*conv_s_b[t]
__global__ __launch_bounds__(256)
void explode_vssum(const float* __restrict__ vpJ, const float* __restrict__ csw,
                   const float* __restrict__ csb, float* __restrict__ vssum) {
  const int idx = blockIdx.x * blockDim.x + threadIdx.x;
  if (idx >= B_ * H_ * FO_ * HS_) return;
  const int c = idx % HS_;
  const int t = (idx / HS_) % FO_;
  const int h = (idx / (HS_ * FO_)) % H_;
  const int b = idx / (HS_ * FO_ * H_);
  float s = 0.f;
#pragma unroll
  for (int f = 0; f < F_; ++f)
    s += csw[t * F_ + f] * vpJ[((size_t)b * F_ + f) * 256 + h * 32 + c];
  vssum[idx] = s + (float)J_ * csb[t];
}

// diag[((b*H+h)*FO+t)*J+s] = softmax_u( mask ? qs[s]*ku[u] : NEG )[u=s]
__global__ __launch_bounds__(256)
void softmax_diag_kernel(const float* __restrict__ qs2d, const float* __restrict__ ku,
                         const int* __restrict__ mask_s, float* __restrict__ diag) {
  const int idx = blockIdx.x * blockDim.x + threadIdx.x;
  if (idx >= B_ * H_ * FO_ * J_) return;
  const int s = idx % J_;
  const int t = (idx / J_) % FO_;
  const int h = (idx / (J_ * FO_)) % H_;
  const int b = idx / (J_ * FO_ * H_);
  const float qv = qs2d[(((size_t)b * FO_ + t) * J_ + s) * 16 + h];
  const float* kur = ku + (((size_t)b * H_ + h) * FO_ + t) * J_;
  const int* mrow = mask_s + s * J_;
  float z[J_];
  float m = -INFINITY;
#pragma unroll
  for (int u = 0; u < J_; ++u) {
    const float zz = (mrow[u] != 0) ? qv * kur[u] : NEG_;
    z[u] = zz;
    m = fmaxf(m, zz);
  }
  float sum = 0.f;
#pragma unroll
  for (int u = 0; u < J_; ++u) sum += __expf(z[u] - m);
  diag[idx] = __expf(z[s] - m) / sum;
}

// out[b,t,s,h*64+c] = c<32 ? diag*vssum[c] : vtsum[c-32]   (float4 vectorized)
__global__ __launch_bounds__(256)
void write_out_kernel(const float* __restrict__ diag, const float* __restrict__ vssum,
                      const float* __restrict__ vtsum, float4* __restrict__ out) {
  const int idx = blockIdx.x * blockDim.x + threadIdx.x;
  if (idx >= B_ * FO_ * J_ * (OUT_ / 4)) return;
  const int ch4 = idx & 127;           // OUT_/4 = 128
  const int r = idx >> 7;              // (b*FO+t)*J+s
  const int s = r % J_;
  const int bt = r / J_;
  const int t = bt % FO_;
  const int b = bt / FO_;
  const int ch = ch4 << 2;
  const int h = ch >> 6;
  const int cc = ch & 63;
  float4 val;
  if (cc < HS_) {
    const float d = diag[(((size_t)b * H_ + h) * FO_ + t) * J_ + s];
    const float* vp = vssum + (((size_t)b * H_ + h) * FO_ + t) * HS_ + cc;
    val = make_float4(d * vp[0], d * vp[1], d * vp[2], d * vp[3]);
  } else {
    const float* vp = vtsum + ((size_t)b * J_ + s) * 256 + h * HS_ + (cc - HS_);
    val = make_float4(vp[0], vp[1], vp[2], vp[3]);
  }
  out[idx] = val;
}

extern "C" void kernel_launch(void* const* d_in, const int* in_sizes, int n_in,
                              void* d_out, int out_size, void* d_ws, size_t ws_size,
                              hipStream_t stream) {
  (void)in_sizes; (void)n_in; (void)out_size; (void)ws_size;
  const float* q  = (const float*)d_in[0];
  const float* k  = (const float*)d_in[1];
  const float* v  = (const float*)d_in[2];
  const int*  mask_s = (const int*)d_in[3];
  // d_in[4] mask_t, d_in[13] conv_t_w, d_in[14] conv_t_b provably unused:
  // sum over the softmax axis of attn_t is identically 1.
  const float* Wq = (const float*)d_in[5];
  const float* bq = (const float*)d_in[6];
  const float* Wk = (const float*)d_in[7];
  const float* bk = (const float*)d_in[8];
  const float* Wv = (const float*)d_in[9];
  const float* bv = (const float*)d_in[10];
  const float* csw = (const float*)d_in[11];
  const float* csb = (const float*)d_in[12];

  char* ws = (char*)d_ws;
  size_t off = 0;
  auto carve = [&](size_t bytes) -> void* {
    void* p = ws + off;
    off = (off + bytes + 255) & ~(size_t)255;
    return p;
  };
  const int MQ = B_ * FO_ * J_;   // 180224
  const int MK = B_ * F_ * J_;    // 90112
  const int MVJ = B_ * F_;        // 4096
  const int MVF = B_ * J_;        // 5632

  _Float16* WqT  = (_Float16*)carve(16 * 256 * sizeof(_Float16));
  _Float16* WkT  = (_Float16*)carve(16 * 256 * sizeof(_Float16));
  _Float16* WvsT = (_Float16*)carve(256 * 256 * sizeof(_Float16));
  _Float16* WvtT = (_Float16*)carve(256 * 256 * sizeof(_Float16));
  float* biasq   = (float*)carve(16 * sizeof(float));
  float* biask   = (float*)carve(16 * sizeof(float));
  float* bias_vs = (float*)carve(256 * sizeof(float));
  float* bias_vt = (float*)carve(256 * sizeof(float));
  float* qs2d  = (float*)carve((size_t)MQ * 16 * sizeof(float));
  float* ks2d  = (float*)carve((size_t)MK * 16 * sizeof(float));
  float* vJsum = (float*)carve((size_t)MVJ * 256 * sizeof(float));
  float* vFsum = (float*)carve((size_t)MVF * 256 * sizeof(float));
  float* vpJ   = (float*)carve((size_t)MVJ * 256 * sizeof(float));
  float* vpF   = (float*)carve((size_t)MVF * 256 * sizeof(float)); // == vtsum
  float* ku    = (float*)carve((size_t)B_ * H_ * FO_ * J_ * sizeof(float));
  float* vssum = (float*)carve((size_t)B_ * H_ * FO_ * HS_ * sizeof(float));
  float* diag  = (float*)carve((size_t)B_ * H_ * FO_ * J_ * sizeof(float));

  // 1. Weight folding / transposes + raw-v reductions (independent).
  prep_weights<<<64, 256, 0, stream>>>(Wq, bq, Wk, bk, Wv, bv,
                                       WqT, biasq, WkT, biask,
                                       WvsT, bias_vs, WvtT, bias_vt);
  reduce_v_J<<<(B_ * F_ * IN_ + 255) / 256, 256, 0, stream>>>(v, vJsum);
  reduce_v_F<<<(B_ * J_ * IN_ + 255) / 256, 256, 0, stream>>>(v, vFsum);

  // 2. Four WMMA GEMMs (K=256, f32->f16 convert on the fly, f32 accumulate).
  gemm16_wmma<8, 1><<<dim3(MQ / 128, 1), 256, 0, stream>>>(q, WqT, biasq, 1.0f, qs2d, 16);
  gemm16_wmma<8, 1><<<dim3(MK / 128, 1), 256, 0, stream>>>(k, WkT, biask, 1.0f, ks2d, 16);
  gemm16_wmma<2, 4><<<dim3(MVJ / 32, 4), 256, 0, stream>>>(vJsum, WvsT, bias_vs, (float)J_, vpJ, 256);
  gemm16_wmma<2, 4><<<dim3(MVF / 32, 4), 256, 0, stream>>>(vFsum, WvtT, bias_vt, (float)F_, vpF, 256);

  // 3. Tiny conv folds (F=16) for the spatial path.
  explode_ku<<<(B_ * H_ * FO_ * J_ + 255) / 256, 256, 0, stream>>>(ks2d, csw, csb, ku);
  explode_vssum<<<(B_ * H_ * FO_ * HS_ + 255) / 256, 256, 0, stream>>>(vpJ, csw, csb, vssum);

  // 4. Masked softmax diagonal over the rank-1 scores.
  softmax_diag_kernel<<<(B_ * H_ * FO_ * J_ + 255) / 256, 256, 0, stream>>>(qs2d, ku, mask_s, diag);

  // 5. Streaming epilogue: 369 MB output write (the roofline-dominant step).
  write_out_kernel<<<(B_ * FO_ * J_ * (OUT_ / 4) + 255) / 256, 256, 0, stream>>>(
      diag, vssum, vpF, (float4*)d_out);
}
